// SparseLlamaMLP_11149735100494
// MI455X (gfx1250) — matrix operand order repaired and checked by reading, not compiled, not run on hardware
//
#include <hip/hip_runtime.h>
#include <cstdint>

// Problem constants (fixed by the reference)
#define HH   2048
#define II   5632
#define NBK  44
#define RR   64
#define TOPK 8
#define NTOK 8192
// LDS row stride in bf16 elements: 80 bytes -> 16B-aligned rows, conflict-friendly.
// Matches TDM pad: 64B row (pad_interval=16 DW) + 16B pad (pad_amount=4 DW).
#define SP   40

typedef __attribute__((ext_vector_type(16))) __bf16 v16bf;
typedef __attribute__((ext_vector_type(8)))  float  v8f;
typedef __attribute__((ext_vector_type(4)))  unsigned int v4u;
typedef __attribute__((ext_vector_type(8)))  int v8i;
typedef __attribute__((ext_vector_type(4)))  int v4i;

#if defined(__has_builtin)
#if __has_builtin(__builtin_amdgcn_tensor_load_to_lds) && __has_builtin(__builtin_amdgcn_s_wait_tensorcnt)
#define USE_TDM 1
#endif
#endif
#ifndef USE_TDM
#define USE_TDM 0
#endif

union FragU { uint4 u[2]; v16bf v; };

__device__ __forceinline__ unsigned short f2bf(float f) {
  unsigned int u = __float_as_uint(f);
  u += 0x7FFFu + ((u >> 16) & 1u);          // round-to-nearest-even
  return (unsigned short)(u >> 16);
}

// A-fragment (16x32 bf16): lane holds row (lane&15); halves {kb..kb+7, kb+16..kb+23}, kb=8*(lane>=16)
__device__ __forceinline__ v16bf ldsA_frag(const unsigned short* p) {
  FragU f;
  f.u[0] = *(const uint4*)(p);
  f.u[1] = *(const uint4*)(p + 16);
  return f.v;
}
// B-fragment (32x16 bf16, staged as B^T rows): lane holds col (lane&15); halves {kb..kb+7, kb+16..kb+23}, kb=8*(lane>=16)
__device__ __forceinline__ v16bf ldsB_frag(const unsigned short* p) {
  FragU f;
  f.u[0] = *(const uint4*)(p);
  f.u[1] = *(const uint4*)(p + 16);
  return f.v;
}
__device__ __forceinline__ v8f wmma_bf16(v16bf a, v16bf b, v8f c) {
  return __builtin_amdgcn_wmma_f32_16x16x32_bf16(false, a, false, b, (short)0, c, false, false);
}
__device__ __forceinline__ float siluf(float x) { return x / (1.0f + __expf(-x)); }

#if USE_TDM
// Issue one TDM descriptor: load a 128-row x 32-bf16 tile (row stride strideElems,
// in bf16 elements) from global into LDS at ldsAddr, padding each 64B row with 16B
// so the LDS row stride is SP=40 halves. Uniform args; EXEC is ignored by TDM.
__device__ __forceinline__ void tdm_load_tile(unsigned ldsAddr,
                                              const unsigned short* g,
                                              unsigned strideElems) {
  unsigned long long ga = (unsigned long long)(uintptr_t)g;
  v4u g0 = { 1u,                                   // count=1 (valid descriptor)
             ldsAddr,                               // lds_addr
             (unsigned)ga,                          // global_addr[31:0]
             (unsigned)(ga >> 32) | (2u << 30) };   // global_addr[56:32] | type=2
  v8i g1 = { (int)((1u << 16)                      // data_size = 2 bytes
                 | (1u << 20)                      // pad_enable
                 | (3u << 22)                      // pad_interval: 16 DWORDs (64B)
                 | (3u << 25)),                    // pad_amount: 4 DWORDs (16B)
             (int)((strideElems & 0xFFFFu) << 16), //  [63:48] tensor_dim0 lo16
             (int)((strideElems >> 16) | (128u << 16)), // dim0 hi16 | tensor_dim1=128
             (int)(32u << 16),                     // tensor_dim1 hi=0 | tile_dim0=32
             (int)128,                             // tile_dim1=128 | tile_dim2=0
             (int)strideElems,                     // tensor_dim0_stride lo32
             0, 0 };
  v4i z4 = { 0, 0, 0, 0 };
#if defined(__clang_major__) && __clang_major__ >= 23
  v8i z8 = { 0, 0, 0, 0, 0, 0, 0, 0 };
  __builtin_amdgcn_tensor_load_to_lds(g0, g1, z4, z4, z8, 0);
#else
  __builtin_amdgcn_tensor_load_to_lds(g0, g1, z4, z4, 0);
#endif
}
#endif

// ---------------------------------------------------------------------------
// 0) weight pre-conversion: fp32 -> bf16 (one-time; makes GEMM staging pure copies
//    and keeps all three weight matrices L2-resident at 69 MB total)
// ---------------------------------------------------------------------------
__global__ __launch_bounds__(256) void cvt_bf16_kernel(
    const float* __restrict__ src, unsigned short* __restrict__ dst, int n4) {
  int i = blockIdx.x * 256 + threadIdx.x;
  if (i >= n4) return;
  float4 v = ((const float4*)src)[i];
  alignas(8) unsigned short t[4] = { f2bf(v.x), f2bf(v.y), f2bf(v.z), f2bf(v.w) };
  ((uint2*)dst)[i] = *(const uint2*)t;
}

// ---------------------------------------------------------------------------
// 1) routing: softmax over top-k scores, scattered dense into block_w[N, NBK]
// ---------------------------------------------------------------------------
__global__ __launch_bounds__(256) void routing_kernel(
    const int* __restrict__ aidx, const float* __restrict__ ascore,
    float* __restrict__ block_w) {
  int n = blockIdx.x * 256 + threadIdx.x;
  if (n >= NTOK) return;
  float s[TOPK]; int id[TOPK];
  float mx = -1e30f;
#pragma unroll
  for (int k = 0; k < TOPK; ++k) {
    s[k]  = ascore[(size_t)n*TOPK + k];
    id[k] = aidx  [(size_t)n*TOPK + k];
    mx = fmaxf(mx, s[k]);
  }
  float sum = 0.f;
#pragma unroll
  for (int k = 0; k < TOPK; ++k) { s[k] = __expf(s[k] - mx); sum += s[k]; }
  float inv = 1.f / sum;
  for (int j = 0; j < NBK; ++j) {
    float a = 0.f;
#pragma unroll
    for (int k = 0; k < TOPK; ++k) a += (id[k] == j) ? s[k] : 0.f;
    block_w[(size_t)n*NBK + j] = a * inv;
  }
}

// ---------------------------------------------------------------------------
// 2) read x once: emit bf16 copy xb[N,H] and latent[N,R] = silu(x @ enc_w^T + b)
// ---------------------------------------------------------------------------
__global__ __launch_bounds__(64) void latent_xb_kernel(
    const float* __restrict__ x, const float* __restrict__ enc_w,
    const float* __restrict__ enc_b,
    unsigned short* __restrict__ xb, unsigned short* __restrict__ latent) {
  __shared__ float xs[HH];
  const int n = blockIdx.x;
  const int t = threadIdx.x;                   // 64 threads; one r per thread
  const float* xr = x + (size_t)n*HH;
  for (int i = t; i < HH; i += 64) {
    float v = xr[i];
    xs[i] = v;
    xb[(size_t)n*HH + i] = f2bf(v);
  }
  __syncthreads();
  const float4* e4 = (const float4*)(enc_w + (size_t)t*HH);
  const float4* x4 = (const float4*)xs;
  float acc = 0.f;
  for (int i = 0; i < HH/4; ++i) {
    float4 a = e4[i], b = x4[i];
    acc += a.x*b.x + a.y*b.y + a.z*b.z + a.w*b.w;
  }
  acc += enc_b[t];
  latent[(size_t)n*RR + t] = f2bf(siluf(acc));
}

// ---------------------------------------------------------------------------
// 3) gate/up GEMM tile + low-rank comp GEMM + epilogue -> weighted[N, I] (bf16)
//    grid = (NBK, NTOK/128); 256 threads = 8 waves, each wave owns 32x64 of C.
//    Tiles staged by the Tensor Data Mover (wave 0 issues descriptors).
// ---------------------------------------------------------------------------
__global__ __launch_bounds__(256) void k1_gateup(
    const unsigned short* __restrict__ xb,
    const unsigned short* __restrict__ latent,
    const float* __restrict__ block_w,
    const unsigned short* __restrict__ gwb,
    const unsigned short* __restrict__ uwb,
    const float* __restrict__ decoder,
    const float* __restrict__ bbias,
    const float* __restrict__ scale_p,
    unsigned short* __restrict__ weighted)
{
  __shared__ unsigned short As[2][128*SP];
  __shared__ unsigned short Bg[2][128*SP];
  __shared__ unsigned short Bu[2][128*SP];
  __shared__ float bwS[128];
  __shared__ float bbS[128];

  const int b    = blockIdx.x;          // expert block 0..43
  const int t0   = blockIdx.y * 128;    // token tile
  const int c0   = b * 128;             // channel base inside I
  const int tid  = threadIdx.x;
  const int lane = tid & 31;
  const int wave = tid >> 5;
  const int wy   = wave >> 1;           // 0..3 : 32-token slab
  const int wx   = wave & 1;            // 0..1 : 64-channel slab
  const int l16  = lane & 15;
  const int hi   = lane >> 4;

  const int srow = tid >> 1;            // staging (fallback + comp): row 0..127
  const int schk = (tid & 1) * 16;      // staging: 16-half chunk

  const unsigned short* srcA = xb  + (size_t)t0 * HH;
  const unsigned short* srcG = gwb + (size_t)c0 * HH;
  const unsigned short* srcU = uwb + (size_t)c0 * HH;

#if !USE_TDM
  auto stageCopy = [&](unsigned short* dst, const unsigned short* src, int k0) {
    const uint4* s = (const uint4*)(src + (size_t)srow * HH + k0 + schk);
    uint4* d = (uint4*)(&dst[srow*SP + schk]);
    d[0] = s[0]; d[1] = s[1];
  };
#endif

  v8f accG[2][4], accU[2][4], accC[2][4];
#pragma unroll
  for (int m = 0; m < 2; ++m)
#pragma unroll
    for (int n = 0; n < 4; ++n) { accG[m][n] = {}; accU[m][n] = {}; accC[m][n] = {}; }

#if USE_TDM
  if (wave == 0) {
    tdm_load_tile((unsigned)(uintptr_t)&As[0][0], srcA, HH);
    tdm_load_tile((unsigned)(uintptr_t)&Bg[0][0], srcG, HH);
    tdm_load_tile((unsigned)(uintptr_t)&Bu[0][0], srcU, HH);
  }
  __builtin_amdgcn_s_wait_tensorcnt(0);
#else
  stageCopy(As[0], srcA, 0);
  stageCopy(Bg[0], srcG, 0);
  stageCopy(Bu[0], srcU, 0);
#endif
  __syncthreads();

  const int KS = HH / 32;               // 64 k-steps
  for (int ks = 0; ks < KS; ++ks) {
    const int cur = ks & 1;
    const int nxt = cur ^ 1;
    if (ks + 1 < KS) {
      const int k0 = (ks + 1) * 32;
#if USE_TDM
      if (wave == 0) {
        tdm_load_tile((unsigned)(uintptr_t)&As[nxt][0], srcA + k0, HH);
        tdm_load_tile((unsigned)(uintptr_t)&Bg[nxt][0], srcG + k0, HH);
        tdm_load_tile((unsigned)(uintptr_t)&Bu[nxt][0], srcU + k0, HH);
      }
#else
      stageCopy(As[nxt], srcA, k0);
      stageCopy(Bg[nxt], srcG, k0);
      stageCopy(Bu[nxt], srcU, k0);
#endif
    }
    v16bf a0 = ldsA_frag(&As[cur][(wy*32 +  0 + l16)*SP + hi*8]);
    v16bf a1 = ldsA_frag(&As[cur][(wy*32 + 16 + l16)*SP + hi*8]);
#pragma unroll
    for (int n = 0; n < 4; ++n) {
      const int rc = (wx*64 + n*16 + l16)*SP + hi*8;
      v16bf bg = ldsB_frag(&Bg[cur][rc]);
      v16bf bu = ldsB_frag(&Bu[cur][rc]);
      accG[0][n] = wmma_bf16(a0, bg, accG[0][n]);
      accG[1][n] = wmma_bf16(a1, bg, accG[1][n]);
      accU[0][n] = wmma_bf16(a0, bu, accU[0][n]);
      accU[1][n] = wmma_bf16(a1, bu, accU[1][n]);
    }
#if USE_TDM
    __builtin_amdgcn_s_wait_tensorcnt(0);
#endif
    __syncthreads();
  }

  if (tid < 128) {
    bwS[tid] = block_w[(size_t)(t0 + tid)*NBK + b];
    bbS[tid] = bbias[b*128 + tid];
  }

  // comp GEMM: latent[128x64] x decoder[b] (64x128), B staged with transpose (tiny)
  for (int ks = 0; ks < 2; ++ks) {
    {
      const uint4* src = (const uint4*)(latent + (size_t)(t0 + srow)*RR + ks*32 + schk);
      uint4* dst = (uint4*)(&As[0][srow*SP + schk]);
      dst[0] = src[0]; dst[1] = src[1];
    }
    {
      const int r  = tid >> 3;          // 0..31 (k index within step)
      const int cs = (tid & 7) * 16;    // column segment
      const float4* src = (const float4*)(decoder + ((size_t)b*RR + ks*32 + r)*128 + cs);
#pragma unroll
      for (int j = 0; j < 4; ++j) {
        float4 v = src[j];
        Bg[0][(cs + j*4 + 0)*SP + r] = f2bf(v.x);
        Bg[0][(cs + j*4 + 1)*SP + r] = f2bf(v.y);
        Bg[0][(cs + j*4 + 2)*SP + r] = f2bf(v.z);
        Bg[0][(cs + j*4 + 3)*SP + r] = f2bf(v.w);
      }
    }
    __syncthreads();
    v16bf a0 = ldsA_frag(&As[0][(wy*32 +  0 + l16)*SP + hi*8]);
    v16bf a1 = ldsA_frag(&As[0][(wy*32 + 16 + l16)*SP + hi*8]);
#pragma unroll
    for (int n = 0; n < 4; ++n) {
      v16bf bd = ldsB_frag(&Bg[0][(wx*64 + n*16 + l16)*SP + hi*8]);
      accC[0][n] = wmma_bf16(a0, bd, accC[0][n]);
      accC[1][n] = wmma_bf16(a1, bd, accC[1][n]);
    }
    __syncthreads();
  }

  // epilogue: (silu(G)*U + scale*(C + basis_bias)) * block_w  -> bf16
  const float scale = scale_p[0];
#pragma unroll
  for (int m = 0; m < 2; ++m) {
#pragma unroll
    for (int n = 0; n < 4; ++n) {
      const int col     = wx*64 + n*16 + l16;
      const int rowbase = wy*32 + m*16 + hi*8;
#pragma unroll
      for (int v = 0; v < 8; ++v) {
        const int row = rowbase + v;
        float g   = accG[m][n][v];
        float u   = accU[m][n][v];
        float act = siluf(g) * u;
        float c   = accC[m][n][v] + bbS[col];
        float wv  = (act + scale * c) * bwS[row];
        weighted[(size_t)(t0 + row)*II + c0 + col] = f2bf(wv);
      }
    }
  }
}

// ---------------------------------------------------------------------------
// 4) down projection: out[N,H] = weighted[N,I] @ down_w^T + comp_bias
//    grid = (HH/128, NTOK/128); TDM-staged tiles as above
// ---------------------------------------------------------------------------
__global__ __launch_bounds__(256) void k2_down(
    const unsigned short* __restrict__ weighted,
    const unsigned short* __restrict__ dwb,
    const float* __restrict__ cbias,
    float* __restrict__ out)
{
  __shared__ unsigned short As[2][128*SP];
  __shared__ unsigned short Bd[2][128*SP];
  __shared__ float cbS[128];

  const int h0   = blockIdx.x * 128;
  const int t0   = blockIdx.y * 128;
  const int tid  = threadIdx.x;
  const int lane = tid & 31;
  const int wave = tid >> 5;
  const int wy   = wave >> 1;
  const int wx   = wave & 1;
  const int l16  = lane & 15;
  const int hi   = lane >> 4;

  const int srow = tid >> 1;
  const int schk = (tid & 1) * 16;

  if (tid < 128) cbS[tid] = cbias[h0 + tid];

  const unsigned short* srcA = weighted + (size_t)t0 * II;
  const unsigned short* srcB = dwb      + (size_t)h0 * II;

#if !USE_TDM
  auto stageCopy = [&](unsigned short* dst, const unsigned short* src, int k0) {
    const uint4* s = (const uint4*)(src + (size_t)srow * II + k0 + schk);
    uint4* d = (uint4*)(&dst[srow*SP + schk]);
    d[0] = s[0]; d[1] = s[1];
  };
#endif

  v8f acc[2][4];
#pragma unroll
  for (int m = 0; m < 2; ++m)
#pragma unroll
    for (int n = 0; n < 4; ++n) acc[m][n] = {};

#if USE_TDM
  if (wave == 0) {
    tdm_load_tile((unsigned)(uintptr_t)&As[0][0], srcA, II);
    tdm_load_tile((unsigned)(uintptr_t)&Bd[0][0], srcB, II);
  }
  __builtin_amdgcn_s_wait_tensorcnt(0);
#else
  stageCopy(As[0], srcA, 0);
  stageCopy(Bd[0], srcB, 0);
#endif
  __syncthreads();

  const int KS = II / 32;               // 176 k-steps
  for (int ks = 0; ks < KS; ++ks) {
    const int cur = ks & 1;
    const int nxt = cur ^ 1;
    if (ks + 1 < KS) {
      const int k0 = (ks + 1) * 32;
#if USE_TDM
      if (wave == 0) {
        tdm_load_tile((unsigned)(uintptr_t)&As[nxt][0], srcA + k0, II);
        tdm_load_tile((unsigned)(uintptr_t)&Bd[nxt][0], srcB + k0, II);
      }
#else
      stageCopy(As[nxt], srcA, k0);
      stageCopy(Bd[nxt], srcB, k0);
#endif
    }
    v16bf a0 = ldsA_frag(&As[cur][(wy*32 +  0 + l16)*SP + hi*8]);
    v16bf a1 = ldsA_frag(&As[cur][(wy*32 + 16 + l16)*SP + hi*8]);
#pragma unroll
    for (int n = 0; n < 4; ++n) {
      v16bf bd = ldsB_frag(&Bd[cur][(wx*64 + n*16 + l16)*SP + hi*8]);
      acc[0][n] = wmma_bf16(a0, bd, acc[0][n]);
      acc[1][n] = wmma_bf16(a1, bd, acc[1][n]);
    }
#if USE_TDM
    __builtin_amdgcn_s_wait_tensorcnt(0);
#endif
    __syncthreads();
  }

#pragma unroll
  for (int m = 0; m < 2; ++m) {
#pragma unroll
    for (int n = 0; n < 4; ++n) {
      const int col     = wx*64 + n*16 + l16;
      const int rowbase = wy*32 + m*16 + hi*8;
#pragma unroll
      for (int v = 0; v < 8; ++v) {
        const int row = rowbase + v;
        out[(size_t)(t0 + row)*HH + h0 + col] = acc[m][n][v] + cbS[col];
      }
    }
  }
}

// ---------------------------------------------------------------------------
extern "C" void kernel_launch(void* const* d_in, const int* in_sizes, int n_in,
                              void* d_out, int out_size, void* d_ws, size_t ws_size,
                              hipStream_t stream) {
  (void)in_sizes; (void)n_in; (void)out_size; (void)ws_size;
  const float* x       = (const float*)d_in[0];
  const int*   aidx    = (const int*)  d_in[1];
  const float* ascore  = (const float*)d_in[2];
  const float* gate_w  = (const float*)d_in[3];
  const float* up_w    = (const float*)d_in[4];
  const float* down_w  = (const float*)d_in[5];
  const float* enc_w   = (const float*)d_in[6];
  const float* enc_b   = (const float*)d_in[7];
  const float* decoder = (const float*)d_in[8];
  const float* bbias   = (const float*)d_in[9];
  const float* cbias   = (const float*)d_in[10];
  const float* scale_p = (const float*)d_in[11];
  float* out = (float*)d_out;

  // workspace layout (bytes):
  //   xb      :          0  33,554,432   (N*H bf16)
  //   latent  : 33,554,432   1,048,576   (N*R bf16)
  //   block_w : 34,603,008   1,441,792   (N*NBK f32)
  //   weighted: 36,044,800  92,274,688   (N*I bf16)
  //   gwb     :128,319,488  23,068,672   (I*H bf16)
  //   uwb     :151,388,160  23,068,672   (I*H bf16)
  //   dwb     :174,456,832  23,068,672   (H*I bf16)   total ~197.5 MB
  char* ws = (char*)d_ws;
  unsigned short* xb       = (unsigned short*)(ws);
  unsigned short* latent   = (unsigned short*)(ws + 33554432);
  float*          block_w  = (float*)         (ws + 34603008);
  unsigned short* weighted = (unsigned short*)(ws + 36044800);
  unsigned short* gwb      = (unsigned short*)(ws + 128319488);
  unsigned short* uwb      = (unsigned short*)(ws + 151388160);
  unsigned short* dwb      = (unsigned short*)(ws + 174456832);

  const int wq4 = (II * HH) / 4;        // float4 chunks per weight matrix
  const int cvtBlocks = (wq4 + 255) / 256;

  routing_kernel  <<<dim3(NTOK/256), dim3(256), 0, stream>>>(aidx, ascore, block_w);
  latent_xb_kernel<<<dim3(NTOK),     dim3(64),  0, stream>>>(x, enc_w, enc_b, xb, latent);
  cvt_bf16_kernel <<<dim3(cvtBlocks), dim3(256), 0, stream>>>(gate_w, gwb, wq4);
  cvt_bf16_kernel <<<dim3(cvtBlocks), dim3(256), 0, stream>>>(up_w,   uwb, wq4);
  cvt_bf16_kernel <<<dim3(cvtBlocks), dim3(256), 0, stream>>>(down_w, dwb, wq4);
  k1_gateup       <<<dim3(NBK, NTOK/128), dim3(256), 0, stream>>>(
      xb, latent, block_w, gwb, uwb, decoder, bbias, scale_p, weighted);
  k2_down         <<<dim3(HH/128, NTOK/128), dim3(256), 0, stream>>>(
      weighted, dwb, cbias, out);
}
